// GeneratorWithMPN_352187318322
// MI455X (gfx1250) — matrix operand order repaired and checked by reading, not compiled, not run on hardware
//
#include <hip/hip_runtime.h>
#include <hip/hip_bf16.h>

typedef __attribute__((ext_vector_type(16))) __bf16 v16bf;
typedef __attribute__((ext_vector_type(8)))  float  v8f;
typedef __attribute__((ext_vector_type(16))) unsigned short v16u;
typedef __attribute__((ext_vector_type(8)))  unsigned short v8u;

__device__ __forceinline__ unsigned short f2bf(float x) {
  union { float f; unsigned u; } v; v.f = x;
  unsigned u = v.u;
  u += 0x7FFFu + ((u >> 16) & 1u);          // round-to-nearest-even
  return (unsigned short)(u >> 16);
}
__device__ __forceinline__ int imin(int a, int b) { return a < b ? a : b; }
__device__ __forceinline__ int imax(int a, int b) { return a > b ? a : b; }

// ---------------------------------------------------------------------------
// Implicit-GEMM 3x3 conv (pad=1, stride=1), NCHW fp32 in/out, bf16 WMMA math.
// Block tile: 64 Cout x 64 contiguous pixels of one image row. 8 waves,
// each wave owns one 16(Cout) x 32(pixels) C-tile (two 16x16 accumulators).
// Requires W % 64 == 0 (true for all layers routed here).
// ---------------------------------------------------------------------------
__global__ __launch_bounds__(256) void k_conv3x3_wmma(
    const float* __restrict__ in, const float* __restrict__ wgt,
    const float* __restrict__ bias, float* __restrict__ out,
    int Cin, int Cout, int H, int W, int relu, int out_coff, int out_ctot)
{
  __shared__ __align__(32) unsigned short sIn[3 * 68 * 32];   // [row][col][k]
  __shared__ __align__(32) unsigned short sW [9 * 64 * 32];   // [tap][m][k]

  const int HW  = H * W;
  const int b   = blockIdx.z;
  const int m0  = blockIdx.y << 6;
  const int n0  = blockIdx.x << 6;
  const int h   = n0 / W;
  const int w0  = n0 % W;
  const int tid  = threadIdx.x;
  const int lane = tid & 31;
  const int wave = tid >> 5;
  const int mi   = wave >> 1;            // 0..3  (Cout sub-tile)
  const int ni   = wave & 1;             // 0..1  (pixel sub-tile)
  const int hi16 = (lane >> 4) & 1;

  v8f acc0 = {};
  v8f acc1 = {};

  const int mIdx  = (mi << 4) + (lane & 15);
  const int koffA = hi16 ? 16 : 0;       // A frag: k = i + koffA
  const int khiB  = hi16 ? 16 : 0;       // B frag: k = i + khiB
  const int ncol  = (ni << 5) + (lane & 15);

  for (int c0 = 0; c0 < Cin; c0 += 32) {
    // -- stage input patch: rows h-1..h+1, cols w0-1..w0+64, 32 channels --
    for (int idx = tid; idx < 3 * 66 * 32; idx += 256) {
      int k   = idx & 31;
      int t   = idx >> 5;
      int col = t % 66;
      int r   = t / 66;
      int hh  = h - 1 + r;
      int ww  = w0 - 1 + col;
      int ch  = c0 + k;
      float v = 0.f;
      if ((unsigned)hh < (unsigned)H && (unsigned)ww < (unsigned)W && ch < Cin)
        v = in[((size_t)(b * Cin + ch)) * HW + (size_t)hh * W + ww];
      sIn[(r * 68 + col) * 32 + k] = f2bf(v);
    }
    // -- stage weights: 9 taps x 64 Cout x 32 Cin --
    for (int idx = tid; idx < 9 * 64 * 32; idx += 256) {
      int k   = idx & 31;
      int t   = idx >> 5;
      int m   = t & 63;
      int tap = t >> 6;
      int co  = m0 + m;
      int ch  = c0 + k;
      float v = 0.f;
      if (co < Cout && ch < Cin)
        v = wgt[((size_t)co * Cin + ch) * 9 + tap];
      sW[(tap * 64 + m) * 32 + k] = f2bf(v);
    }
    __syncthreads();

#pragma unroll
    for (int tap = 0; tap < 9; ++tap) {
      const int kh = tap / 3;
      const int kw = tap - 3 * kh;
      const unsigned short* ap = &sW[(tap * 64 + mIdx) * 32 + koffA];
      v8u alo = *(const v8u*)(ap);
      v8u ahi = *(const v8u*)(ap + 8);
      v16u araw = __builtin_shufflevector(alo, ahi,
          0, 1, 2, 3, 4, 5, 6, 7, 8, 9, 10, 11, 12, 13, 14, 15);
      v16bf A = __builtin_bit_cast(v16bf, araw);
      const int colb = ncol + kw;
      v16u b0 = *(const v16u*)(&sIn[(kh * 68 + colb) * 32 + khiB]);
      v16u b1 = *(const v16u*)(&sIn[(kh * 68 + colb + 16) * 32 + khiB]);
      acc0 = __builtin_amdgcn_wmma_f32_16x16x32_bf16(
          false, A, false, __builtin_bit_cast(v16bf, b0), (short)0, acc0, false, false);
      acc1 = __builtin_amdgcn_wmma_f32_16x16x32_bf16(
          false, A, false, __builtin_bit_cast(v16bf, b1), (short)0, acc1, false, false);
    }
    __syncthreads();
  }

  // -- epilogue: C/D layout -> (cout, pixel), fused bias + ReLU --
  const int nf    = lane & 15;
  const int cbase = m0 + (mi << 4) + (hi16 ? 8 : 0);
  const int pos0  = n0 + (ni << 5) + nf;
#pragma unroll
  for (int j = 0; j < 8; ++j) {
    int co = cbase + j;
    if (co < Cout) {
      float bb = bias[co];
      float v0 = acc0[j] + bb;
      float v1 = acc1[j] + bb;
      if (relu) { v0 = fmaxf(v0, 0.f); v1 = fmaxf(v1, 0.f); }
      size_t obase = ((size_t)b * out_ctot + out_coff + co) * HW;
      out[obase + pos0]      = v0;
      out[obase + pos0 + 16] = v1;
    }
  }
}

// ------------------------- memory-bound scalar kernels ----------------------

__global__ void k_sobel(const float* __restrict__ in, float* __restrict__ out,
                        int BC, int H, int W) {
  int idx = blockIdx.x * blockDim.x + threadIdx.x;
  int total = BC * H * W;
  if (idx >= total) return;
  int w = idx % W, h = (idx / W) % H, c = idx / (W * H);
  const float* p = in + (size_t)c * H * W;
  auto at = [&](int y, int x) -> float {
    if ((unsigned)y >= (unsigned)H || (unsigned)x >= (unsigned)W) return 0.f;
    return p[y * W + x];
  };
  float a00 = at(h - 1, w - 1), a01 = at(h - 1, w), a02 = at(h - 1, w + 1);
  float a10 = at(h, w - 1),                        a12 = at(h, w + 1);
  float a20 = at(h + 1, w - 1), a21 = at(h + 1, w), a22 = at(h + 1, w + 1);
  float gx = a00 - a02 + 2.f * (a10 - a12) + a20 - a22;
  float gy = a00 + 2.f * a01 + a02 - a20 - 2.f * a21 - a22;
  out[idx] = sqrtf(gx * gx + gy * gy + 1e-6f);
}

__global__ void k_wavelet(const float* __restrict__ in, float* __restrict__ out,
                          int B, int C, int H, int W) {
  int idx = blockIdx.x * blockDim.x + threadIdx.x;
  int total = B * C * 5 * H * W;
  if (idx >= total) return;
  int w = idx % W, h = (idx / W) % H;
  int oc = (idx / (H * W)) % (5 * C);
  int b = idx / (H * W * 5 * C);
  int c = oc / 5, band = oc % 5;
  const float* p = in + ((size_t)b * C + c) * H * W;
  float v;
  if (band == 0) {
    v = p[h * W + w];
  } else {
    int H2 = H / 2, W2 = W / 2;
    float sy = 0.5f * h - 0.25f, sx = 0.5f * w - 0.25f;
    int y0 = (int)floorf(sy); float fy = sy - y0;
    int x0 = (int)floorf(sx); float fx = sx - x0;
    auto bandval = [&](int i, int j) -> float {
      i = imin(imax(i, 0), H2 - 1);
      j = imin(imax(j, 0), W2 - 1);
      float a  = p[(2 * i) * W + 2 * j];
      float bb = p[(2 * i) * W + 2 * j + 1];
      float cc = p[(2 * i + 1) * W + 2 * j];
      float dd = p[(2 * i + 1) * W + 2 * j + 1];
      if (band == 1) return (a + bb + cc + dd) * 0.5f;
      if (band == 2) return (a + bb - cc - dd) * 0.5f;
      if (band == 3) return (a - bb + cc - dd) * 0.5f;
      return (a - bb - cc + dd) * 0.5f;
    };
    float top = bandval(y0, x0) * (1.f - fx) + bandval(y0, x0 + 1) * fx;
    float bot = bandval(y0 + 1, x0) * (1.f - fx) + bandval(y0 + 1, x0 + 1) * fx;
    v = top * (1.f - fy) + bot * fy;
  }
  out[idx] = v;
}

__global__ void k_dwconv(const float* __restrict__ in, const float* __restrict__ wgt,
                         const float* __restrict__ bias, float* __restrict__ out,
                         int B, int C, int H, int W) {
  int idx = blockIdx.x * blockDim.x + threadIdx.x;
  int total = B * C * H * W;
  if (idx >= total) return;
  int w = idx % W, h = (idx / W) % H, c = (idx / (H * W)) % C;
  const float* p = in + (size_t)(idx / (H * W)) * H * W;
  const float* wk = wgt + c * 9;
  float sum = 0.f;
  for (int ky = 0; ky < 3; ++ky)
    for (int kx = 0; kx < 3; ++kx) {
      int y = h + ky - 1, x = w + kx - 1;
      if ((unsigned)y < (unsigned)H && (unsigned)x < (unsigned)W)
        sum += p[y * W + x] * wk[ky * 3 + kx];
    }
  out[idx] = sum + bias[c];
}

__global__ void k_conv1x1(const float* __restrict__ in, const float* __restrict__ wgt,
                          const float* __restrict__ bias, float* __restrict__ out,
                          int B, int Cin, int Cout, int HW, int relu,
                          int coff, int ctot) {
  int idx = blockIdx.x * blockDim.x + threadIdx.x;
  int total = B * Cout * HW;
  if (idx >= total) return;
  int pos = idx % HW;
  int co = (idx / HW) % Cout;
  int b = idx / (HW * Cout);
  float sum = bias[co];
  for (int ci = 0; ci < Cin; ++ci)
    sum += in[((size_t)b * Cin + ci) * HW + pos] * wgt[co * Cin + ci];
  if (relu) sum = fmaxf(sum, 0.f);
  out[((size_t)b * ctot + coff + co) * HW + pos] = sum;
}

__global__ void k_conv3x3_scalar(const float* __restrict__ in, const float* __restrict__ wgt,
                                 const float* __restrict__ bias, float* __restrict__ out,
                                 int B, int Cin, int Cout, int H, int W, int relu) {
  int idx = blockIdx.x * blockDim.x + threadIdx.x;
  int total = B * Cout * H * W;
  if (idx >= total) return;
  int w = idx % W, h = (idx / W) % H;
  int co = (idx / (H * W)) % Cout;
  int b = idx / (H * W * Cout);
  float sum = bias[co];
  for (int ci = 0; ci < Cin; ++ci) {
    const float* p = in + ((size_t)b * Cin + ci) * H * W;
    const float* wk = wgt + ((size_t)co * Cin + ci) * 9;
    for (int ky = 0; ky < 3; ++ky)
      for (int kx = 0; kx < 3; ++kx) {
        int y = h + ky - 1, x = w + kx - 1;
        if ((unsigned)y < (unsigned)H && (unsigned)x < (unsigned)W)
          sum += p[y * W + x] * wk[ky * 3 + kx];
      }
  }
  if (relu) sum = fmaxf(sum, 0.f);
  out[idx] = sum;
}

__global__ void k_tile_msg(const int* __restrict__ bm, float* __restrict__ out, int B) {
  int idx = blockIdx.x * blockDim.x + threadIdx.x;
  int total = B * 3 * 256;
  if (idx >= total) return;
  int pos = idx % 256;
  int b = idx / (256 * 3);
  out[idx] = (float)bm[b * 256 + pos];
}

__global__ void k_pool(const float* __restrict__ in, float* __restrict__ out,
                       int B, int C, int H, int W, int in_coff, int in_ctot) {
  int Ho = H / 2, Wo = W / 2;
  int idx = blockIdx.x * blockDim.x + threadIdx.x;
  int total = B * C * Ho * Wo;
  if (idx >= total) return;
  int wo = idx % Wo, ho = (idx / Wo) % Ho;
  int c = (idx / (Ho * Wo)) % C;
  int b = idx / (Ho * Wo * C);
  const float* p = in + ((size_t)b * in_ctot + in_coff + c) * H * W;
  int h2 = 2 * ho, w2 = 2 * wo;
  float m = fmaxf(fmaxf(p[h2 * W + w2], p[h2 * W + w2 + 1]),
                  fmaxf(p[(h2 + 1) * W + w2], p[(h2 + 1) * W + w2 + 1]));
  out[idx] = m;
}

// half-pixel bilinear (jax.image.resize 'bilinear')
__global__ void k_resize_half(const float* __restrict__ in, float* __restrict__ out,
                              int B, int C, int Hi, int Wi, int Ho, int Wo,
                              int coff, int ctot) {
  int idx = blockIdx.x * blockDim.x + threadIdx.x;
  int total = B * C * Ho * Wo;
  if (idx >= total) return;
  int w = idx % Wo, h = (idx / Wo) % Ho;
  int c = (idx / (Ho * Wo)) % C;
  int b = idx / (Ho * Wo * C);
  float sy = (h + 0.5f) * (float)Hi / (float)Ho - 0.5f;
  float sx = (w + 0.5f) * (float)Wi / (float)Wo - 0.5f;
  int y0 = (int)floorf(sy); float fy = sy - y0;
  int x0 = (int)floorf(sx); float fx = sx - x0;
  int y0c = imin(imax(y0, 0), Hi - 1), y1c = imin(imax(y0 + 1, 0), Hi - 1);
  int x0c = imin(imax(x0, 0), Wi - 1), x1c = imin(imax(x0 + 1, 0), Wi - 1);
  const float* p = in + ((size_t)b * C + c) * Hi * Wi;
  float v = (p[y0c * Wi + x0c] * (1.f - fx) + p[y0c * Wi + x1c] * fx) * (1.f - fy)
          + (p[y1c * Wi + x0c] * (1.f - fx) + p[y1c * Wi + x1c] * fx) * fy;
  out[((size_t)b * ctot + coff + c) * Ho * Wo + (size_t)h * Wo + w] = v;
}

// align-corners bilinear (the reference's custom _up_ac)
__global__ void k_resize_ac(const float* __restrict__ in, float* __restrict__ out,
                            int B, int C, int Hi, int Wi, int f,
                            int coff, int ctot) {
  int Ho = Hi * f, Wo = Wi * f;
  int idx = blockIdx.x * blockDim.x + threadIdx.x;
  int total = B * C * Ho * Wo;
  if (idx >= total) return;
  int w = idx % Wo, h = (idx / Wo) % Ho;
  int c = (idx / (Ho * Wo)) % C;
  int b = idx / (Ho * Wo * C);
  float sy = (float)h * (float)(Hi - 1) / (float)(Ho - 1);
  float sx = (float)w * (float)(Wi - 1) / (float)(Wo - 1);
  int y0 = (int)floorf(sy); int y1 = imin(y0 + 1, Hi - 1); float wy = sy - y0;
  int x0 = (int)floorf(sx); int x1 = imin(x0 + 1, Wi - 1); float wx = sx - x0;
  const float* p = in + ((size_t)b * C + c) * Hi * Wi;
  float top = p[y0 * Wi + x0] * (1.f - wx) + p[y0 * Wi + x1] * wx;
  float bot = p[y1 * Wi + x0] * (1.f - wx) + p[y1 * Wi + x1] * wx;
  out[((size_t)b * ctot + coff + c) * Ho * Wo + (size_t)h * Wo + w] =
      top * (1.f - wy) + bot * wy;
}

// ---------------------------------------------------------------------------

extern "C" void kernel_launch(void* const* d_in, const int* in_sizes, int n_in,
                              void* d_out, int out_size, void* d_ws, size_t ws_size,
                              hipStream_t stream) {
  (void)in_sizes; (void)n_in; (void)out_size; (void)ws_size;
  const float* cover  = (const float*)d_in[0];
  const int*   bmsg   = (const int*)d_in[1];
  const float* dw_w   = (const float*)d_in[2];
  const float* proj_w = (const float*)d_in[3];
  const float* mpn_w1 = (const float*)d_in[4];
  const float* mpn_w2 = (const float*)d_in[5];
  const float* e1w1   = (const float*)d_in[6];
  const float* e1w2   = (const float*)d_in[7];
  const float* e2w1   = (const float*)d_in[8];
  const float* e2w2   = (const float*)d_in[9];
  const float* e3w1   = (const float*)d_in[10];
  const float* e3w2   = (const float*)d_in[11];
  const float* fw1    = (const float*)d_in[12];
  const float* fw2    = (const float*)d_in[13];
  const float* dw_b   = (const float*)d_in[14];
  const float* proj_b = (const float*)d_in[15];
  const float* mpn_b1 = (const float*)d_in[16];
  const float* mpn_b2 = (const float*)d_in[17];
  const float* e1b1   = (const float*)d_in[18];
  const float* e1b2   = (const float*)d_in[19];
  const float* e2b1   = (const float*)d_in[20];
  const float* e2b2   = (const float*)d_in[21];
  const float* e3b1   = (const float*)d_in[22];
  const float* e3b2   = (const float*)d_in[23];
  const float* fb1    = (const float*)d_in[24];
  const float* fb2    = (const float*)d_in[25];
  float* outp = (float*)d_out;

  const int B = 4, H = 256, W = 256, HW = H * W;

  float* ws = (float*)d_ws;
  size_t off = 0;
  auto alloc = [&](size_t n) { float* p = ws + off; off += n; return p; };
  float* sob    = alloc((size_t)B * 3 * HW);
  float* wav    = alloc((size_t)B * 15 * HW);
  float* dep    = alloc((size_t)B * 15 * HW);
  float* fused  = alloc((size_t)B * 64 * HW);          // [0:32)=image_feat [32:64)=msg_up
  float* inner  = alloc((size_t)B * 64 * HW);          // shared scratch for inner convs
  float* xc     = alloc((size_t)B * 448 * HW);         // [0:64)=x1 [64:192)=up2 [192:448)=up4
  float* p1     = alloc((size_t)B * 64 * (HW / 4));
  float* x2     = alloc((size_t)B * 128 * (HW / 4));
  float* p2     = alloc((size_t)B * 128 * (HW / 16));
  float* x3     = alloc((size_t)B * 256 * (HW / 16));
  float* msg    = alloc((size_t)B * 3 * 256);
  float* msob   = alloc((size_t)B * 3 * 256);
  float* mwav   = alloc((size_t)B * 15 * 256);
  float* mdep   = alloc((size_t)B * 15 * 256);
  float* minner = alloc((size_t)B * 64 * 256);
  float* mfeat  = alloc((size_t)B * 32 * 256);

  dim3 blk(256);
  auto g1 = [](long long n) { return dim3((unsigned)((n + 255) / 256)); };

  // ---- image feature path ----
  k_sobel  <<<g1((long long)B * 3 * HW),  blk, 0, stream>>>(cover, sob, B * 3, H, W);
  k_wavelet<<<g1((long long)B * 15 * HW), blk, 0, stream>>>(sob, wav, B, 3, H, W);
  k_dwconv <<<g1((long long)B * 15 * HW), blk, 0, stream>>>(wav, dw_w, dw_b, dep, B, 15, H, W);
  k_conv1x1<<<g1((long long)B * 32 * HW), blk, 0, stream>>>(dep, proj_w, proj_b, fused,
                                                            B, 15, 32, HW, 0, 0, 64);
  // ---- message path (16x16, tiny -> scalar) ----
  k_tile_msg<<<g1(B * 3 * 256),  blk, 0, stream>>>(bmsg, msg, B);
  k_sobel   <<<g1(B * 3 * 256),  blk, 0, stream>>>(msg, msob, B * 3, 16, 16);
  k_wavelet <<<g1(B * 15 * 256), blk, 0, stream>>>(msob, mwav, B, 3, 16, 16);
  k_dwconv  <<<g1(B * 15 * 256), blk, 0, stream>>>(mwav, dw_w, dw_b, mdep, B, 15, 16, 16);
  k_conv3x3_scalar<<<g1(B * 64 * 256), blk, 0, stream>>>(mdep, mpn_w1, mpn_b1, minner,
                                                         B, 15, 64, 16, 16, 1);
  k_conv3x3_scalar<<<g1(B * 32 * 256), blk, 0, stream>>>(minner, mpn_w2, mpn_b2, mfeat,
                                                         B, 64, 32, 16, 16, 1);
  k_resize_half<<<g1((long long)B * 32 * HW), blk, 0, stream>>>(mfeat, fused,
                                                  B, 32, 16, 16, H, W, 32, 64);
  // ---- encoder (WMMA implicit GEMM) ----
  k_conv3x3_wmma<<<dim3(HW / 64, 1, B), blk, 0, stream>>>(fused, e1w1, e1b1, inner,
                                                          64, 64, H, W, 1, 0, 64);
  k_conv3x3_wmma<<<dim3(HW / 64, 1, B), blk, 0, stream>>>(inner, e1w2, e1b2, xc,
                                                          64, 64, H, W, 1, 0, 448);
  k_pool<<<g1((long long)B * 64 * (HW / 4)), blk, 0, stream>>>(xc, p1, B, 64, H, W, 0, 448);
  k_conv3x3_wmma<<<dim3((HW / 4) / 64, 2, B), blk, 0, stream>>>(p1, e2w1, e2b1, inner,
                                                          64, 128, H / 2, W / 2, 1, 0, 128);
  k_conv3x3_wmma<<<dim3((HW / 4) / 64, 2, B), blk, 0, stream>>>(inner, e2w2, e2b2, x2,
                                                          128, 128, H / 2, W / 2, 1, 0, 128);
  k_pool<<<g1((long long)B * 128 * (HW / 16)), blk, 0, stream>>>(x2, p2, B, 128,
                                                          H / 2, W / 2, 0, 128);
  k_conv3x3_wmma<<<dim3((HW / 16) / 64, 4, B), blk, 0, stream>>>(p2, e3w1, e3b1, inner,
                                                          128, 256, H / 4, W / 4, 1, 0, 256);
  k_conv3x3_wmma<<<dim3((HW / 16) / 64, 4, B), blk, 0, stream>>>(inner, e3w2, e3b2, x3,
                                                          256, 256, H / 4, W / 4, 1, 0, 256);
  // ---- upsample into concat slices ----
  k_resize_ac<<<g1((long long)B * 128 * HW), blk, 0, stream>>>(x2, xc, B, 128,
                                                          H / 2, W / 2, 2, 64, 448);
  k_resize_ac<<<g1((long long)B * 256 * HW), blk, 0, stream>>>(x3, xc, B, 256,
                                                          H / 4, W / 4, 4, 192, 448);
  // ---- fusion head ----
  k_conv3x3_wmma<<<dim3(HW / 64, 1, B), blk, 0, stream>>>(xc, fw1, fb1, inner,
                                                          448, 64, H, W, 1, 0, 64);
  k_conv1x1<<<g1((long long)B * 3 * HW), blk, 0, stream>>>(inner, fw2, fb2, outp,
                                                          B, 64, 3, HW, 0, 0, 3);
}
